// CollisionOperatorBGK_30236569764428
// MI455X (gfx1250) — compile-verified, matrix operand order checked by reading, not run on hardware
//
#include <hip/hip_runtime.h>

typedef __attribute__((ext_vector_type(16))) _Float16 v16h;
typedef __attribute__((ext_vector_type(8)))  float    v8f;

#define QPTS 64
#define DXI (70.0f / 63.0f)
#define MLP_WAVES 4

// ---------------------------------------------------------------------------
// Branch-free fast tanh: tanh(x) = sign(x) * (1 - e) / (1 + e), e = exp(-2|x|)
// ---------------------------------------------------------------------------
__device__ __forceinline__ float fast_tanh(float x) {
  float e = __expf(-2.0f * fabsf(x));
  float r = (1.0f - e) * __builtin_amdgcn_rcpf(1.0f + e);
  return copysignf(r, x);
}

// ---------------------------------------------------------------------------
// Kernel 1: Newton-Raphson for lambda (exp-family equilibrium), mu0 = -log S0.
// Uses geometric recurrence r^k (1 exp + FMA chain) instead of 64 exps/iter.
// ---------------------------------------------------------------------------
__global__ void __launch_bounds__(256)
newton_kernel(const float* __restrict__ macro, float* __restrict__ lam_out,
              float* __restrict__ mu0_out, int n) {
  int i = blockIdx.x * blockDim.x + threadIdx.x;
  if (i >= n) return;
  float v = macro[2 * i];
  float lam = 0.0f;
#pragma unroll 1
  for (int it = 0; it < 100; ++it) {
    float r = __expf(lam * DXI);
    float p = 1.0f, S0 = 0.0f, S1 = 0.0f, S2 = 0.0f;
#pragma unroll 8
    for (int k = 0; k < QPTS; ++k) {
      float kf = (float)k;
      S0 += p;
      S1 = fmaf(kf, p, S1);
      S2 = fmaf(kf * kf, p, S2);
      p *= r;
    }
    float m1  = DXI * (S1 / S0);
    float var = fmaf(DXI * DXI, S2 / S0, -m1 * m1);
    var = fmaxf(var, 1e-20f);
    float step = (m1 - v) / var;
    lam -= step;
    if (fabsf(step) < 1e-9f) break;
  }
  float r = __expf(lam * DXI);
  float p = 1.0f, S0 = 0.0f;
#pragma unroll 8
  for (int k = 0; k < QPTS; ++k) { S0 += p; p *= r; }
  lam_out[i] = lam;
  mu0_out[i] = -__logf(S0);
}

// ---------------------------------------------------------------------------
// One 16-row dense layer: Y[16x64] = tanh(X[16x64] @ W^T + b), via WMMA.
// WSTRIDE: row stride of W in LDS; KCHUNKS: number of 32-wide K chunks.
// ---------------------------------------------------------------------------
template <int WSTRIDE, int KCHUNKS>
__device__ __forceinline__ void layer_wmma(const _Float16* __restrict__ Xbuf,
                                           const _Float16* __restrict__ Wl,
                                           const float*    __restrict__ bias,
                                           _Float16*       __restrict__ Ybuf,
                                           int g, int nl) {
#pragma unroll
  for (int t = 0; t < 4; ++t) {
    union { v8f f; float s[8]; } c;
    float bv = bias[t * 16 + nl];
#pragma unroll
    for (int v = 0; v < 8; ++v) c.s[v] = bv;
#pragma unroll
    for (int kc = 0; kc < KCHUNKS; ++kc) {
      union { v16h h; unsigned u[8]; } a, b;
#pragma unroll
      for (int v = 0; v < 8; ++v) {
        // wave32 16-bit A/B fragment packing: pair (k0, k0+1) per VGPR
        int k0 = 2 * (v & 3) + 8 * g + 16 * (v >> 2) + 32 * kc;
        a.u[v] = *(const unsigned*)&Xbuf[nl * 64 + k0];
        b.u[v] = *(const unsigned*)&Wl[(t * 16 + nl) * WSTRIDE + k0];
      }
      c.f = __builtin_amdgcn_wmma_f32_16x16x32_f16(
          false, a.h, false, b.h, (short)0, c.f, false, false);
    }
    // C layout: M = v + 8g, N = t*16 + nl
#pragma unroll
    for (int v = 0; v < 8; ++v)
      Ybuf[(v + 8 * g) * 64 + t * 16 + nl] = (_Float16)fast_tanh(c.s[v]);
  }
}

// ---------------------------------------------------------------------------
// Kernel 2: tau MLP via f16 WMMA. One wave = one 16-row tile.
// ---------------------------------------------------------------------------
__global__ void __launch_bounds__(32 * MLP_WAVES)
mlp_tau_kernel(const float* __restrict__ macro, const float* __restrict__ pos,
               const float* __restrict__ W0, const float* __restrict__ b0,
               const float* __restrict__ W1, const float* __restrict__ b1,
               const float* __restrict__ W2, const float* __restrict__ b2,
               const float* __restrict__ W3, const float* __restrict__ b3,
               const float* __restrict__ W4, const float* __restrict__ b4,
               float* __restrict__ tau_out, int n) {
  __shared__ _Float16 sW0[64 * 32];                 // layer 0, K zero-padded to 32
  __shared__ _Float16 sW[3][64 * 64];               // W1..W3, [out][in]
  __shared__ float    sB[4][64];                    // b0..b3
  __shared__ float    sW4[64];                      // final layer weights
  __shared__ _Float16 sX[MLP_WAVES][2][16 * 64];    // per-wave ping/pong tiles

  const int tid = threadIdx.x;

  // ---- cooperative weight staging (f32 -> f16) ----
  for (int idx = tid; idx < 64 * 32; idx += blockDim.x) {
    int o = idx >> 5, k = idx & 31;
    sW0[idx] = (k < 3) ? (_Float16)W0[o * 3 + k] : (_Float16)0.0f;
  }
  {
    const float* Ws[3] = {W1, W2, W3};
    for (int l = 0; l < 3; ++l)
      for (int idx = tid; idx < 64 * 64; idx += blockDim.x)
        sW[l][idx] = (_Float16)Ws[l][idx];
    const float* Bs[4] = {b0, b1, b2, b3};
    for (int l = 0; l < 4; ++l)
      for (int idx = tid; idx < 64; idx += blockDim.x)
        sB[l][idx] = Bs[l][idx];
    for (int idx = tid; idx < 64; idx += blockDim.x)
      sW4[idx] = W4[idx];
  }

  const int  wave  = tid >> 5;
  const int  lane  = tid & 31;
  const int  g     = lane >> 4;   // half-wave group
  const int  nl    = lane & 15;   // row (A/C) / col (B/C) index within tile
  const long tile0 = (long)blockIdx.x * (MLP_WAVES * 16) + wave * 16;

  // ---- stage input tile [16 x 64], cols 0..2 = features, rest zero ----
  {
    _Float16* X0 = sX[wave][0];
    for (int idx = lane; idx < 16 * 64; idx += 32) {
      int r = idx >> 6, c = idx & 63;
      long node = tile0 + r; if (node >= n) node = n - 1;
      float val = 0.0f;
      if (c == 0)      val = macro[2 * node];
      else if (c == 1) val = macro[2 * node + 1];
      else if (c == 2) val = pos[node];
      X0[idx] = (_Float16)val;
    }
  }
  __syncthreads();

  // ---- layer 0 (K padded to 32) ----
  layer_wmma<32, 1>(sX[wave][0], sW0, sB[0], sX[wave][1], g, nl);
  __syncthreads();

  // ---- layers 1..3 (K = 64) ----
  int cur = 1;
#pragma unroll 1
  for (int L = 1; L < 4; ++L) {
    layer_wmma<64, 2>(sX[wave][cur], sW[L - 1], sB[L], sX[wave][cur ^ 1], g, nl);
    cur ^= 1;
    __syncthreads();
  }

  // ---- final 64 -> 1 layer + exp; both half-waves split each row's dot ----
  {
    const _Float16* Xf = sX[wave][cur];
    float acc = 0.0f;
#pragma unroll
    for (int k = 0; k < 32; ++k)
      acc = fmaf((float)Xf[nl * 64 + 32 * g + k], sW4[32 * g + k], acc);
    acc += __shfl_xor(acc, 16, 32);
    if (g == 0) {
      long node = tile0 + nl;
      if (node < n) tau_out[node] = __expf(acc + b4[0]);
    }
  }
}

// ---------------------------------------------------------------------------
// Kernel 3: omega = (f_eq - f) / tau. One wave per node row (Q=64, 2/lane).
// Bandwidth-bound: 128 MB read + 128 MB write -> ~11 us at 23.3 TB/s.
// ---------------------------------------------------------------------------
__global__ void __launch_bounds__(256)
omega_kernel(const float* __restrict__ f, const float* __restrict__ lam,
             const float* __restrict__ mu0, const float* __restrict__ tau,
             float* __restrict__ out, int n) {
  long wave = ((long)blockIdx.x * blockDim.x + threadIdx.x) >> 5;
  int  lane = threadIdx.x & 31;
  if (wave >= n) return;
  const float2* frow = (const float2*)(f + wave * QPTS);
  float2 fv = frow[lane];
  float s = fv.x + fv.y;
#pragma unroll
  for (int m = 16; m >= 1; m >>= 1) s += __shfl_xor(s, m, 32);
  float rho  = s * (1.0f / (float)QPTS);
  float La   = lam[wave];
  float Mu   = mu0[wave];
  float invt = 1.0f / tau[wave];
  float x0 = (float)(2 * lane) * DXI;
  float feq0 = rho * __expf(fmaf(La, x0, Mu));
  float feq1 = rho * __expf(fmaf(La, x0 + DXI, Mu));
  float2 o;
  o.x = (feq0 - fv.x) * invt;
  o.y = (feq1 - fv.y) * invt;
  ((float2*)(out + wave * QPTS))[lane] = o;
}

// ---------------------------------------------------------------------------
extern "C" void kernel_launch(void* const* d_in, const int* in_sizes, int n_in,
                              void* d_out, int out_size, void* d_ws, size_t ws_size,
                              hipStream_t stream) {
  const float* f   = (const float*)d_in[0];
  const float* mf  = (const float*)d_in[1];
  const float* pos = (const float*)d_in[2];
  const float* W0  = (const float*)d_in[3];  const float* b0 = (const float*)d_in[4];
  const float* W1  = (const float*)d_in[5];  const float* b1 = (const float*)d_in[6];
  const float* W2  = (const float*)d_in[7];  const float* b2 = (const float*)d_in[8];
  const float* W3  = (const float*)d_in[9];  const float* b3 = (const float*)d_in[10];
  const float* W4  = (const float*)d_in[11]; const float* b4 = (const float*)d_in[12];

  const int N = in_sizes[1] / 2;  // macro_features is [N,2]

  float* lam = (float*)d_ws;      // [N]
  float* mu0 = lam + N;           // [N]
  float* tau = mu0 + N;           // [N]

  newton_kernel<<<(N + 255) / 256, 256, 0, stream>>>(mf, lam, mu0, N);

  const int rows_per_block = MLP_WAVES * 16;
  mlp_tau_kernel<<<(N + rows_per_block - 1) / rows_per_block, 32 * MLP_WAVES, 0, stream>>>(
      mf, pos, W0, b0, W1, b1, W2, b2, W3, b3, W4, b4, tau, N);

  long threads = (long)N * 32;    // one wave32 per node row
  omega_kernel<<<(unsigned)((threads + 255) / 256), 256, 0, stream>>>(
      f, lam, mu0, tau, (float*)d_out, N);
}
